// GCNModel_1039382086073
// MI455X (gfx1250) — matrix operand order repaired and checked by reading, not compiled, not run on hardware
//
#include <hip/hip_runtime.h>
#include <hip/hip_bf16.h>

#define N_NODES  150000   // multiple of 16 (9375 tiles)
#define N_EDGES  600000
#define N_GRAPHS 4096     // multiple of 16
#define DIN      74
#define HDIM     200
#define NLAYERS  5
#define BN_EPS   1e-5f

#define KPAD_EMB 96       // DIN=74 padded to 3*32
#define KPAD_H   224      // HDIM=200 padded to 7*32
#define NPAD_H   208      // HDIM=200 padded to 13*16
#define NPAD_O1  1024     // already multiple of 16

typedef __attribute__((ext_vector_type(16))) __bf16 v16bf;
typedef __attribute__((ext_vector_type(8)))  float  v8f;
typedef unsigned short u16;
typedef unsigned int   u32;

union FragU { v16bf v; u32 u[8]; };

// fp32 -> bf16 (round-to-nearest-even)
static __device__ __forceinline__ u16 f2bf(float a) {
    u32 u = __builtin_bit_cast(u32, a);
    u += 0x7FFFu + ((u >> 16) & 1u);
    return (u16)(u >> 16);
}

// A fragment (16x32 bf16): row panel in memory, contiguous K run of 16 (32 bytes).
// pA = row start (bf16*), b0 = kb + (lane>=16 ? 16 : 0).
static __device__ __forceinline__ v16bf load_a_frag(const u16* __restrict__ pA, int b0) {
    const uint4 lo = *(const uint4*)(pA + b0);
    const uint4 hi = *(const uint4*)(pA + b0 + 8);
    FragU f;
    f.u[0] = lo.x; f.u[1] = lo.y; f.u[2] = lo.z; f.u[3] = lo.w;
    f.u[4] = hi.x; f.u[5] = hi.y; f.u[6] = hi.z; f.u[7] = hi.w;
    return f.v;
}

// B fragment (32x16 bf16) from column-major weights Wt[Npad][Kpad]:
// lane holds col n; elems 0..15 = contiguous K run of 16 (32 bytes).
// pB = col start, ks = kb + (lane>=16 ? 16 : 0).
static __device__ __forceinline__ v16bf load_b_frag(const u16* __restrict__ pB, int ks) {
    const uint4 lo = *(const uint4*)(pB + ks);
    const uint4 hi = *(const uint4*)(pB + ks + 8);
    FragU f;
    f.u[0] = lo.x; f.u[1] = lo.y; f.u[2] = lo.z; f.u[3] = lo.w;
    f.u[4] = hi.x; f.u[5] = hi.y; f.u[6] = hi.z; f.u[7] = hi.w;
    return f.v;
}

// out = [relu](A @ Wt^T + bias); one wave owns a 16-row panel, A cached in
// registers across all N tiles; B loads double-buffered across K chunks.
// Optionally dual-stores a padded bf16 copy.
template <int KCH, bool RELU, bool WBF>
__global__ void gemm_panel(const u16* __restrict__ A, const u16* __restrict__ Wt,
                           const float* __restrict__ bias, float* __restrict__ out,
                           u16* __restrict__ obf, int M, int Ncols, int Npad, int obfPitch) {
    const int lane = threadIdx.x & 31;
    const int wave = threadIdx.x >> 5;
    const int tm = blockIdx.x * 8 + wave;
    if (tm >= (M >> 4)) return;              // wave-uniform: EXEC all-ones at WMMA
    const int  Kpad = KCH * 32;
    const bool hi = lane >= 16;
    const int  nlane = lane & 15;
    const int  offA = hi ? 16 : 0;
    const int  offB = hi ? 16 : 0;

    const u16* pA = A + (size_t)(tm * 16 + nlane) * Kpad;
    v16bf apan[KCH];
#pragma unroll
    for (int c = 0; c < KCH; ++c) apan[c] = load_a_frag(pA, c * 32 + offA);

    const int tilesN = Npad >> 4;
    for (int tn = 0; tn < tilesN; ++tn) {
        const u16* pB = Wt + (size_t)(tn * 16 + nlane) * Kpad;
        v8f acc = {};
        v16bf bcur = load_b_frag(pB, offB);
#pragma unroll
        for (int c = 0; c < KCH; ++c) {
            v16bf bnext = {};
            if (c + 1 < KCH) bnext = load_b_frag(pB, (c + 1) * 32 + offB);
            acc = __builtin_amdgcn_wmma_f32_16x16x32_bf16(false, apan[c], false, bcur,
                                                          (short)0, acc, false, false);
            bcur = bnext;
        }
        const int col = tn * 16 + nlane;
        if (col < Ncols) {
            const float bv = bias[col];
#pragma unroll
            for (int r = 0; r < 8; ++r) {
                const int m = tm * 16 + (hi ? 8 + r : r);
                float v = acc[r] + bv;
                if (RELU) v = fmaxf(v, 0.f);
                out[(size_t)m * Ncols + col] = v;
                if (WBF) obf[(size_t)m * obfPitch + col] = f2bf(v);
            }
        }
    }
}

// h = relu(aggbf @ Wc^T + bc) + relu(xbf @ Wr^T + br); both A panels
// register-cached, both B streams double-buffered.
template <int KCH>
__global__ void layer_gemm_fused(const u16* __restrict__ Abf, const u16* __restrict__ Xbf,
                                 const u16* __restrict__ Wc, const float* __restrict__ bc,
                                 const u16* __restrict__ Wr, const float* __restrict__ br,
                                 float* __restrict__ h, int M, int Ncols, int Npad) {
    const int lane = threadIdx.x & 31;
    const int wave = threadIdx.x >> 5;
    const int tm = blockIdx.x * 8 + wave;
    if (tm >= (M >> 4)) return;
    const int  Kpad = KCH * 32;
    const bool hi = lane >= 16;
    const int  nlane = lane & 15;
    const int  offA = hi ? 16 : 0;
    const int  offB = hi ? 16 : 0;

    const u16* pA = Abf + (size_t)(tm * 16 + nlane) * Kpad;
    const u16* pX = Xbf + (size_t)(tm * 16 + nlane) * Kpad;
    v16bf apanA[KCH], apanX[KCH];
#pragma unroll
    for (int c = 0; c < KCH; ++c) {
        apanA[c] = load_a_frag(pA, c * 32 + offA);
        apanX[c] = load_a_frag(pX, c * 32 + offA);
    }

    const int tilesN = Npad >> 4;
    for (int tn = 0; tn < tilesN; ++tn) {
        const u16* pBc = Wc + (size_t)(tn * 16 + nlane) * Kpad;
        const u16* pBr = Wr + (size_t)(tn * 16 + nlane) * Kpad;
        v8f accC = {}, accR = {};
        v16bf bCcur = load_b_frag(pBc, offB);
        v16bf bRcur = load_b_frag(pBr, offB);
#pragma unroll
        for (int c = 0; c < KCH; ++c) {
            v16bf bCnext = {}, bRnext = {};
            if (c + 1 < KCH) {
                bCnext = load_b_frag(pBc, (c + 1) * 32 + offB);
                bRnext = load_b_frag(pBr, (c + 1) * 32 + offB);
            }
            accC = __builtin_amdgcn_wmma_f32_16x16x32_bf16(false, apanA[c], false, bCcur,
                                                           (short)0, accC, false, false);
            accR = __builtin_amdgcn_wmma_f32_16x16x32_bf16(false, apanX[c], false, bRcur,
                                                           (short)0, accR, false, false);
            bCcur = bCnext;
            bRcur = bRnext;
        }
        const int col = tn * 16 + nlane;
        if (col < Ncols) {
            const float bvc = bc[col], bvr = br[col];
#pragma unroll
            for (int r = 0; r < 8; ++r) {
                const int m = tm * 16 + (hi ? 8 + r : r);
                float vc = fmaxf(accC[r] + bvc, 0.f);
                float vr = fmaxf(accR[r] + bvr, 0.f);
                h[(size_t)m * Ncols + col] = vc + vr;
            }
        }
    }
}

// Weight prep: W[K x N] fp32 row-major -> Wt[Npad x Kpad] bf16 (zero padded).
__global__ void wt_cvt(const float* __restrict__ W, u16* __restrict__ Wt,
                       int K, int N, int Kpad, int Npad) {
    int idx = blockIdx.x * blockDim.x + threadIdx.x;
    if (idx >= Npad * Kpad) return;
    int n = idx / Kpad, k = idx % Kpad;
    Wt[idx] = (n < N && k < K) ? f2bf(W[(size_t)k * N + n]) : (u16)0;
}

// Row convert: src[M x C] fp32 -> dst[M x Cpad] bf16 (zero padded).
__global__ void cvt_rows(const float* __restrict__ src, u16* __restrict__ dst,
                         long long total, int C, int Cpad) {
    long long idx = (long long)blockIdx.x * blockDim.x + threadIdx.x;
    if (idx >= total) return;
    int c = (int)(idx % Cpad);
    long long r = idx / Cpad;
    dst[idx] = (c < C) ? f2bf(src[r * C + c]) : (u16)0;
}

// agg[dst] += x[src], per-(edge, 4-channel) thread, float4 gathers.
__global__ void scatter_add_edges(const float* __restrict__ x, const int* __restrict__ src,
                                  const int* __restrict__ dst, float* __restrict__ agg) {
    const int C4 = HDIM / 4;
    long long idx = (long long)blockIdx.x * blockDim.x + threadIdx.x;
    if (idx >= (long long)N_EDGES * C4) return;
    int e = (int)(idx / C4);
    int c = (int)(idx % C4) * 4;
    int s = src[e], d = dst[e];
    const float4 v = *(const float4*)(x + (size_t)s * HDIM + c);
    float* p = agg + (size_t)d * HDIM + c;
    atomicAdd(p + 0, v.x); atomicAdd(p + 1, v.y);
    atomicAdd(p + 2, v.z); atomicAdd(p + 3, v.w);
}

// Per-channel sum / sum-of-squares partials via LDS, flushed with global atomics.
__global__ void bn_partial(const float* __restrict__ h, float* __restrict__ sums) {
    __shared__ float ls[2 * HDIM];
    for (int i = threadIdx.x; i < 2 * HDIM; i += blockDim.x) ls[i] = 0.f;
    __syncthreads();
    const int CH = 128;
    long long base = (long long)blockIdx.x * CH * HDIM;
    long long total = (long long)N_NODES * HDIM;
    int count = (int)((base + (long long)CH * HDIM <= total) ? (long long)CH * HDIM
                                                             : (total - base));
    for (int i = threadIdx.x; i < count; i += blockDim.x) {
        float v = h[base + i];
        int c = i % HDIM;
        atomicAdd(&ls[c], v);
        atomicAdd(&ls[HDIM + c], v * v);
    }
    __syncthreads();
    for (int i = threadIdx.x; i < 2 * HDIM; i += blockDim.x) atomicAdd(&sums[i], ls[i]);
}

__global__ void bn_finalize(const float* __restrict__ sums, const float* __restrict__ gamma,
                            const float* __restrict__ beta, float* __restrict__ params) {
    int c = blockIdx.x * blockDim.x + threadIdx.x;
    if (c >= HDIM) return;
    const float inv = 1.f / (float)N_NODES;
    float mean = sums[c] * inv;
    float var  = sums[HDIM + c] * inv - mean * mean;
    float sc   = gamma[c] * rsqrtf(var + BN_EPS);
    params[c]        = sc;
    params[HDIM + c] = beta[c] - mean * sc;
}

// x = h*scale + shift, dual-store fp32 + padded bf16.
__global__ void bn_apply(const float* __restrict__ h, const float* __restrict__ params,
                         float* __restrict__ x, u16* __restrict__ xbf) {
    long long idx = (long long)blockIdx.x * blockDim.x + threadIdx.x;
    if (idx >= (long long)N_NODES * HDIM) return;
    int c = (int)(idx % HDIM);
    long long r = idx / HDIM;
    float v = h[idx] * params[c] + params[HDIM + c];
    x[idx] = v;
    xbf[r * KPAD_H + c] = f2bf(v);
}

__global__ void pool_nodes(const float* __restrict__ x, const int* __restrict__ gid,
                           float* __restrict__ pooled) {
    const int C4 = HDIM / 4;
    long long idx = (long long)blockIdx.x * blockDim.x + threadIdx.x;
    if (idx >= (long long)N_NODES * C4) return;
    int n = (int)(idx / C4);
    int c = (int)(idx % C4) * 4;
    int g = gid[n];
    const float4 v = *(const float4*)(x + (size_t)n * HDIM + c);
    float* p = pooled + (size_t)g * HDIM + c;
    atomicAdd(p + 0, v.x); atomicAdd(p + 1, v.y);
    atomicAdd(p + 2, v.z); atomicAdd(p + 3, v.w);
}

// out[g] = t[g,:] . W2 + b2, one wave per graph, shuffle reduction.
__global__ void readout2(const float* __restrict__ t, const float* __restrict__ W2,
                         const float* __restrict__ b2, float* __restrict__ out) {
    int gw = (blockIdx.x * blockDim.x + threadIdx.x) >> 5;
    int lane = threadIdx.x & 31;
    if (gw >= N_GRAPHS) return;
    const float* row = t + (size_t)gw * 1024;
    float s = 0.f;
    for (int j = lane; j < 1024; j += 32) s += row[j] * W2[j];
#pragma unroll
    for (int o = 16; o > 0; o >>= 1) s += __shfl_down(s, o, 32);
    if (lane == 0) out[gw] = s + b2[0];
}

extern "C" void kernel_launch(void* const* d_in, const int* in_sizes, int n_in,
                              void* d_out, int out_size, void* d_ws, size_t ws_size,
                              hipStream_t stream) {
    (void)in_sizes; (void)n_in; (void)out_size; (void)ws_size;
    const float* node_feat = (const float*)d_in[0];
    const int*   src       = (const int*)d_in[1];
    const int*   dst       = (const int*)d_in[2];
    const int*   gid       = (const int*)d_in[3];
    const float* emb_W     = (const float*)d_in[4];
    const float* emb_b     = (const float*)d_in[5];
    const float* conv_W    = (const float*)d_in[6];
    const float* conv_b    = (const float*)d_in[7];
    const float* res_W     = (const float*)d_in[8];
    const float* res_b     = (const float*)d_in[9];
    const float* bn_g      = (const float*)d_in[10];
    const float* bn_b      = (const float*)d_in[11];
    const float* out_W1    = (const float*)d_in[12];
    const float* out_b1    = (const float*)d_in[13];
    const float* out_W2    = (const float*)d_in[14];
    const float* out_b2    = (const float*)d_in[15];
    float* out = (float*)d_out;

    char* ws = (char*)d_ws;
    size_t off = 0;
    auto carve = [&](size_t bytes) -> void* {
        void* p = (void*)(ws + off);
        off += (bytes + 255) & ~(size_t)255;
        return p;
    };
    float* x       = (float*)carve((size_t)N_NODES * HDIM * 4);
    float* agg     = (float*)carve((size_t)N_NODES * HDIM * 4);      // also reused as h
    u16*   x_bf    = (u16*)  carve((size_t)N_NODES * KPAD_H * 2);
    u16*   agg_bf  = (u16*)  carve((size_t)N_NODES * KPAD_H * 2);
    u16*   nf_bf   = (u16*)  carve((size_t)N_NODES * KPAD_EMB * 2);
    float* pooled  = (float*)carve((size_t)N_GRAPHS * HDIM * 4);
    u16*   pool_bf = (u16*)  carve((size_t)N_GRAPHS * KPAD_H * 2);
    float* t       = (float*)carve((size_t)N_GRAPHS * 1024 * 4);
    float* sums    = (float*)carve(2 * HDIM * 4);
    float* params  = (float*)carve(2 * HDIM * 4);
    u16*   embWt   = (u16*)  carve((size_t)NPAD_H * KPAD_EMB * 2);
    u16*   convWt  = (u16*)  carve((size_t)NLAYERS * NPAD_H * KPAD_H * 2);
    u16*   resWt   = (u16*)  carve((size_t)NLAYERS * NPAD_H * KPAD_H * 2);
    u16*   W1t     = (u16*)  carve((size_t)NPAD_O1 * KPAD_H * 2);

    // ---- weight / input prep (bf16, transposed, zero-padded) ----
    {
        int n = NPAD_H * KPAD_EMB;
        wt_cvt<<<(n + 255) / 256, 256, 0, stream>>>(emb_W, embWt, DIN, HDIM, KPAD_EMB, NPAD_H);
        n = NPAD_H * KPAD_H;
        for (int l = 0; l < NLAYERS; ++l) {
            wt_cvt<<<(n + 255) / 256, 256, 0, stream>>>(
                conv_W + (size_t)l * HDIM * HDIM, convWt + (size_t)l * NPAD_H * KPAD_H,
                HDIM, HDIM, KPAD_H, NPAD_H);
            wt_cvt<<<(n + 255) / 256, 256, 0, stream>>>(
                res_W + (size_t)l * HDIM * HDIM, resWt + (size_t)l * NPAD_H * KPAD_H,
                HDIM, HDIM, KPAD_H, NPAD_H);
        }
        n = NPAD_O1 * KPAD_H;
        wt_cvt<<<(n + 255) / 256, 256, 0, stream>>>(out_W1, W1t, HDIM, 1024, KPAD_H, NPAD_O1);

        long long tot = (long long)N_NODES * KPAD_EMB;
        cvt_rows<<<(int)((tot + 255) / 256), 256, 0, stream>>>(node_feat, nf_bf, tot, DIN, KPAD_EMB);
    }
    // x_bf padding columns stay zero forever (GEMM/bn_apply only write cols < HDIM).
    hipMemsetAsync(x_bf, 0, (size_t)N_NODES * KPAD_H * 2, stream);

    const int mblocksN = ((N_NODES / 16) + 7) / 8;   // 9375 tiles -> 1172 blocks

    // x = node_feat @ emb_W + emb_b  (dual-store fp32 + bf16)
    gemm_panel<KPAD_EMB / 32, false, true><<<mblocksN, 256, 0, stream>>>(
        nf_bf, embWt, emb_b, x, x_bf, N_NODES, HDIM, NPAD_H, KPAD_H);

    const long long sthreads = (long long)N_EDGES * (HDIM / 4);
    const long long ethreads = (long long)N_NODES * HDIM;
    const long long cthreads = (long long)N_NODES * KPAD_H;
    for (int l = 0; l < NLAYERS; ++l) {
        hipMemsetAsync(agg, 0, (size_t)N_NODES * HDIM * 4, stream);
        scatter_add_edges<<<(int)((sthreads + 255) / 256), 256, 0, stream>>>(x, src, dst, agg);
        cvt_rows<<<(int)((cthreads + 255) / 256), 256, 0, stream>>>(agg, agg_bf, cthreads,
                                                                    HDIM, KPAD_H);
        // h (stored into agg fp32) = relu(agg@Wc+bc) + relu(x@Wr+br)
        layer_gemm_fused<KPAD_H / 32><<<mblocksN, 256, 0, stream>>>(
            agg_bf, x_bf,
            convWt + (size_t)l * NPAD_H * KPAD_H, conv_b + (size_t)l * HDIM,
            resWt  + (size_t)l * NPAD_H * KPAD_H, res_b  + (size_t)l * HDIM,
            agg, N_NODES, HDIM, NPAD_H);
        hipMemsetAsync(sums, 0, 2 * HDIM * 4, stream);
        bn_partial<<<(N_NODES + 127) / 128, 256, 0, stream>>>(agg, sums);
        bn_finalize<<<1, 256, 0, stream>>>(sums, bn_g + (size_t)l * HDIM,
                                           bn_b + (size_t)l * HDIM, params);
        bn_apply<<<(int)((ethreads + 255) / 256), 256, 0, stream>>>(agg, params, x, x_bf);
    }

    hipMemsetAsync(pooled, 0, (size_t)N_GRAPHS * HDIM * 4, stream);
    const long long pthreads = (long long)N_NODES * (HDIM / 4);
    pool_nodes<<<(int)((pthreads + 255) / 256), 256, 0, stream>>>(x, gid, pooled);

    const long long gthreads = (long long)N_GRAPHS * KPAD_H;
    cvt_rows<<<(int)((gthreads + 255) / 256), 256, 0, stream>>>(pooled, pool_bf, gthreads,
                                                                HDIM, KPAD_H);
    const int mblocksG = ((N_GRAPHS / 16) + 7) / 8;  // 256 tiles -> 32 blocks
    gemm_panel<KPAD_H / 32, true, false><<<mblocksG, 256, 0, stream>>>(
        pool_bf, W1t, out_b1, t, (u16*)nullptr, N_GRAPHS, 1024, NPAD_O1, 0);

    readout2<<<(N_GRAPHS * 32 + 255) / 256, 256, 0, stream>>>(t, out_W2, out_b2, out);
}
